// MEGNetBlock_53549652246920
// MI455X (gfx1250) — compile-verified, matrix operand order checked by reading, not run on hardware
//
#include <hip/hip_runtime.h>

#define NSITES  100000
#define NBONDS  800000
#define NGRAPHS 1000
#define DF      64

typedef __attribute__((ext_vector_type(16))) __bf16 v16bf;
typedef __attribute__((ext_vector_type(4)))  __bf16 v4bf;
typedef __attribute__((ext_vector_type(8)))  float  v8f;

__device__ __forceinline__ void atomAddF(float* p, float v) {
  unsafeAtomicAdd(p, v);   // global_atomic_add_f32 on gfx1250
}

// CDNA5 async DMA: copy 16B global -> LDS (per lane), tracked by ASYNCcnt.
__device__ __forceinline__ void async_ld16(unsigned lds_off, const void* gaddr) {
  asm volatile("global_load_async_to_lds_b128 %0, %1, off"
               :: "v"(lds_off), "v"(gaddr) : "memory");
}
__device__ __forceinline__ void wait_async() {
  asm volatile("s_wait_asynccnt 0" ::: "memory");
}
__device__ __forceinline__ unsigned lds_off_of(const void* p) {
  return (unsigned)(size_t)p;   // flat LDS aperture: low 32 bits = LDS offset
}

// K index inside a 32-wide K slab for vector element e, lane-half h
// (CDNA5 ISA 7.12.2, 16-bit A-matrix 16x32 layout; B mirrors it with N=lane&15)
__device__ __forceinline__ int frag_k(int e, int half) {
  int v = e >> 1, lo = e & 1;
  return ((v >= 4) ? 16 : 0) + half * 8 + (v & 3) * 2 + lo;
}

template<int K>
__device__ __forceinline__ v16bf load_a_frag(const __bf16* A, int kbase, int lane) {
  int m = lane & 15, half = lane >> 4;
  v16bf a;
#pragma unroll
  for (int e = 0; e < 16; ++e) a[e] = A[m * K + kbase + frag_k(e, half)];
  return a;
}

// Packed weights: [kt][nt][lane][16] bf16, each lane's fragment is one 32B run.
template<int M>
__device__ __forceinline__ v16bf load_b_packed(const __bf16* __restrict__ Wp,
                                               int kt, int nt, int lane) {
  return *reinterpret_cast<const v16bf*>(
      Wp + ((((size_t)kt * (M / 16) + nt) * 32 + lane) << 4));
}

// Y(16xM) = relu(A(16xK) @ W(KxM) + bias), bf16 result into LDS
template<int K, int M>
__device__ __forceinline__ void layer_to_lds(const __bf16* A, const __bf16* __restrict__ Wp,
                                             const float* bias, __bf16* out, int lane) {
  int n = lane & 15, half = lane >> 4;
#pragma unroll
  for (int nt = 0; nt < M / 16; ++nt) {
    v8f acc = {};
#pragma unroll
    for (int kt = 0; kt < K / 32; ++kt) {
      v16bf a = load_a_frag<K>(A, kt * 32, lane);
      v16bf b = load_b_packed<M>(Wp, kt, nt, lane);
      acc = __builtin_amdgcn_wmma_f32_16x16x32_bf16(false, a, false, b, (short)0, acc, false, false);
    }
    float bv = bias[nt * 16 + n];
#pragma unroll
    for (int r = 0; r < 8; ++r) {
      float v = acc[r] + bv;
      out[(r + half * 8) * M + nt * 16 + n] = (__bf16)(v > 0.f ? v : 0.f);
    }
  }
}

// Same, but keep fp32 relu result in registers: outv[nt*8+r] is (row r+8*half, col nt*16+(lane&15))
template<int K, int M>
__device__ __forceinline__ void layer_final(const __bf16* A, const __bf16* __restrict__ Wp,
                                            const float* bias, float* outv, int lane) {
  int n = lane & 15;
#pragma unroll
  for (int nt = 0; nt < M / 16; ++nt) {
    v8f acc = {};
#pragma unroll
    for (int kt = 0; kt < K / 32; ++kt) {
      v16bf a = load_a_frag<K>(A, kt * 32, lane);
      v16bf b = load_b_packed<M>(Wp, kt, nt, lane);
      acc = __builtin_amdgcn_wmma_f32_16x16x32_bf16(false, a, false, b, (short)0, acc, false, false);
    }
    float bv = bias[nt * 16 + n];
#pragma unroll
    for (int r = 0; r < 8; ++r) {
      float v = acc[r] + bv;
      outv[nt * 8 + r] = v > 0.f ? v : 0.f;
    }
  }
}

__device__ __forceinline__ v4bf cvt4(float4 f) {
  v4bf h;
  h[0] = (__bf16)f.x; h[1] = (__bf16)f.y; h[2] = (__bf16)f.z; h[3] = (__bf16)f.w;
  return h;
}

__global__ void zero_kernel(float* p, long n) {
  long i = (long)blockIdx.x * blockDim.x + threadIdx.x;
  if (i < n) p[i] = 0.f;
}

// Re-layout a KxM fp32 weight matrix into fragment-ordered bf16.
__global__ void pack_w_kernel(const float* __restrict__ W, __bf16* __restrict__ P,
                              int K, int M) {
  int t = blockIdx.x * blockDim.x + threadIdx.x;
  if (t >= K * M) return;
  int e = t & 15;
  int lane = (t >> 4) & 31;
  int rest = t >> 9;
  int ntiles = M >> 4;
  int nt = rest % ntiles;
  int kt = rest / ntiles;
  int k = kt * 32 + frag_k(e, lane >> 4);
  int n = nt * 16 + (lane & 15);
  P[t] = (__bf16)W[k * M + n];
}

// Y = relu(relu(X@W1+b1)@W2+b2), X: Nx64. Emits fp32 Y (for residuals) + bf16 Ybf (for gathers).
__global__ void pre_mlp_kernel(const float* __restrict__ X,
                               const __bf16* __restrict__ W1, const float* __restrict__ b1,
                               const __bf16* __restrict__ W2, const float* __restrict__ b2,
                               float* __restrict__ Y, __bf16* __restrict__ Ybf, int N) {
  __shared__ __bf16 sA[4][16 * 64];
  __shared__ __bf16 sM[4][16 * 64];
  int wave = threadIdx.x >> 5, lane = threadIdx.x & 31;
  long r0 = ((long)blockIdx.x * 4 + wave) * 16;
  if (r0 >= N) return;
  __bf16* A = sA[wave];
  for (int i = lane; i < 16 * 16; i += 32) {       // float4 granules
    int r = i >> 4, c4 = i & 15;
    long row = r0 + r;
    float4 f = make_float4(0.f, 0.f, 0.f, 0.f);
    if (row < N) f = reinterpret_cast<const float4*>(X + row * 64)[c4];
    *reinterpret_cast<v4bf*>(&A[r * 64 + c4 * 4]) = cvt4(f);
  }
  asm volatile("" ::: "memory");
  layer_to_lds<64, 64>(A, W1, b1, sM[wave], lane);
  float outv[4 * 8];
  layer_final<64, 64>(sM[wave], W2, b2, outv, lane);
  int n = lane & 15, half = lane >> 4;
#pragma unroll
  for (int nt = 0; nt < 4; ++nt)
#pragma unroll
    for (int r = 0; r < 8; ++r) {
      long row = r0 + r + half * 8;
      if (row < N) {
        float v = outv[nt * 8 + r];
        Y[row * 64 + nt * 16 + n] = v;
        Ybf[row * 64 + nt * 16 + n] = (__bf16)v;
      }
    }
}

__global__ void bond_update_kernel(const __bf16* __restrict__ sites_bf,
                                   const __bf16* __restrict__ bonds_bf,
                                   const __bf16* __restrict__ states_bf,
                                   const float* __restrict__ bonds_pre,
                                   const int* __restrict__ idx1, const int* __restrict__ idx2,
                                   const int* __restrict__ g2b,
                                   const __bf16* __restrict__ W1, const float* __restrict__ b1,
                                   const __bf16* __restrict__ W2, const float* __restrict__ b2,
                                   const __bf16* __restrict__ W3, const float* __restrict__ b3,
                                   float* __restrict__ out_bonds,
                                   float* site_accum, float* site_cnt,
                                   float* gb_accum, float* gb_cnt) {
  __shared__ __bf16 sCat[4][16 * 256];
  __shared__ __bf16 sH1[4][16 * 64];
  __shared__ __bf16 sH2[4][16 * 32];
  int wave = threadIdx.x >> 5, lane = threadIdx.x & 31;
  long r0 = ((long)blockIdx.x * 4 + wave) * 16;
  if (r0 >= NBONDS) return;
  __bf16* Cat = sCat[wave];
  if (lane < 16) {  // per-bond segment counts
    long bond = r0 + lane;
    atomAddF(&site_cnt[idx1[bond]], 1.0f);
    atomAddF(&gb_cnt[g2b[bond]], 1.0f);
  }
  // async DMA gather: 4 sources x 16 rows x 128B (bf16 rows) straight into LDS
  unsigned catBase = lds_off_of(Cat);
  int j = lane & 7, rsub = lane >> 3;       // j: 16B chunk in row, rsub: row subgroup
#pragma unroll
  for (int rr = 0; rr < 4; ++rr) {
    int r = rr * 4 + rsub;
    long bond = r0 + r;
    unsigned l = catBase + (unsigned)r * 512 + (unsigned)j * 16;
    async_ld16(l,       sites_bf + (long)idx1[bond] * 64 + j * 8);
    async_ld16(l + 128, sites_bf + (long)idx2[bond] * 64 + j * 8);
    async_ld16(l + 256, bonds_bf + bond * 64 + j * 8);
    async_ld16(l + 384, states_bf + (long)g2b[bond] * 64 + j * 8);
  }
  wait_async();
  asm volatile("" ::: "memory");
  layer_to_lds<256, 64>(Cat, W1, b1, sH1[wave], lane);
  layer_to_lds<64, 32>(sH1[wave], W2, b2, sH2[wave], lane);
  float outv[4 * 8];
  layer_final<32, 64>(sH2[wave], W3, b3, outv, lane);
  int n = lane & 15, half = lane >> 4;
#pragma unroll
  for (int nt = 0; nt < 4; ++nt)
#pragma unroll
    for (int r = 0; r < 8; ++r) {
      long bond = r0 + r + half * 8;
      int col = nt * 16 + n;
      float v = outv[nt * 8 + r];
      out_bonds[bond * 64 + col] = v + bonds_pre[bond * 64 + col];  // fp32 residual
      atomAddF(&site_accum[(long)idx1[bond] * 64 + col], v);
      atomAddF(&gb_accum[(long)g2b[bond] * 64 + col], v);
    }
}

__global__ void site_update_kernel(const float* __restrict__ sites_pre,
                                   const __bf16* __restrict__ sites_bf,
                                   const __bf16* __restrict__ states_bf,
                                   const float* __restrict__ site_accum,
                                   const float* __restrict__ site_cnt,
                                   const int* __restrict__ g2s,
                                   const __bf16* __restrict__ W1, const float* __restrict__ b1,
                                   const __bf16* __restrict__ W2, const float* __restrict__ b2,
                                   const __bf16* __restrict__ W3, const float* __restrict__ b3,
                                   float* __restrict__ out_sites,
                                   float* gs_accum, float* gs_cnt) {
  __shared__ __bf16 sCat[4][16 * 192];
  __shared__ __bf16 sH1[4][16 * 64];
  __shared__ __bf16 sH2[4][16 * 32];
  int wave = threadIdx.x >> 5, lane = threadIdx.x & 31;
  long r0 = ((long)blockIdx.x * 4 + wave) * 16;
  if (r0 >= NSITES) return;
  __bf16* Cat = sCat[wave];
  if (lane < 16) {  // per-site graph counts
    long s = r0 + lane;
    atomAddF(&gs_cnt[g2s[s]], 1.0f);
  }
  // async DMA: cols 64..191 (sites_bf row, states_bf gather row)
  unsigned catBase = lds_off_of(Cat);
  int j = lane & 7, rsub = lane >> 3;
#pragma unroll
  for (int rr = 0; rr < 4; ++rr) {
    int r = rr * 4 + rsub;
    long s = r0 + r;
    unsigned l = catBase + (unsigned)r * 384 + (unsigned)j * 16;
    async_ld16(l + 128, sites_bf + s * 64 + j * 8);
    async_ld16(l + 256, states_bf + (long)g2s[s] * 64 + j * 8);
  }
  // cols 0..63: pooled bonds = accum / count (fp32 math, then cvt)
  for (int i = lane; i < 16 * 16; i += 32) {
    int r = i >> 4, c4 = i & 15;
    long s = r0 + r;
    float4 f = reinterpret_cast<const float4*>(site_accum + s * 64)[c4];
    float inv = 1.f / fmaxf(site_cnt[s], 1.f);
    f.x *= inv; f.y *= inv; f.z *= inv; f.w *= inv;
    *reinterpret_cast<v4bf*>(&Cat[r * 192 + c4 * 4]) = cvt4(f);
  }
  wait_async();
  asm volatile("" ::: "memory");
  layer_to_lds<192, 64>(Cat, W1, b1, sH1[wave], lane);
  layer_to_lds<64, 32>(sH1[wave], W2, b2, sH2[wave], lane);
  float outv[4 * 8];
  layer_final<32, 64>(sH2[wave], W3, b3, outv, lane);
  int n = lane & 15, half = lane >> 4;
#pragma unroll
  for (int nt = 0; nt < 4; ++nt)
#pragma unroll
    for (int r = 0; r < 8; ++r) {
      long s = r0 + r + half * 8;
      int col = nt * 16 + n;
      float v = outv[nt * 8 + r];
      out_sites[s * 64 + col] = v + sites_pre[s * 64 + col];  // fp32 residual
      atomAddF(&gs_accum[(long)g2s[s] * 64 + col], v);
    }
}

__global__ void state_update_kernel(const float* __restrict__ states_pre,
                                    const float* __restrict__ gb_accum,
                                    const float* __restrict__ gb_cnt,
                                    const float* __restrict__ gs_accum,
                                    const float* __restrict__ gs_cnt,
                                    const __bf16* __restrict__ W1, const float* __restrict__ b1,
                                    const __bf16* __restrict__ W2, const float* __restrict__ b2,
                                    const __bf16* __restrict__ W3, const float* __restrict__ b3,
                                    float* __restrict__ out_states) {
  __shared__ __bf16 sCat[4][16 * 192];
  __shared__ __bf16 sH1[4][16 * 64];
  __shared__ __bf16 sH2[4][16 * 32];
  int wave = threadIdx.x >> 5, lane = threadIdx.x & 31;
  long r0 = ((long)blockIdx.x * 4 + wave) * 16;
  if (r0 >= NGRAPHS) return;
  __bf16* Cat = sCat[wave];
  for (int i = lane; i < 16 * 48; i += 32) {
    int r = i / 48, c4 = i % 48;
    int c = c4 * 4;
    long g = r0 + r;
    float4 f = make_float4(0.f, 0.f, 0.f, 0.f);
    if (g < NGRAPHS) {
      if (c < 64) {
        f = reinterpret_cast<const float4*>(gb_accum + g * 64)[c4];
        float inv = 1.f / fmaxf(gb_cnt[g], 1.f);
        f.x *= inv; f.y *= inv; f.z *= inv; f.w *= inv;
      } else if (c < 128) {
        f = reinterpret_cast<const float4*>(gs_accum + g * 64)[(c - 64) >> 2];
        float inv = 1.f / fmaxf(gs_cnt[g], 1.f);
        f.x *= inv; f.y *= inv; f.z *= inv; f.w *= inv;
      } else {
        f = reinterpret_cast<const float4*>(states_pre + g * 64)[(c - 128) >> 2];
      }
    }
    *reinterpret_cast<v4bf*>(&Cat[r * 192 + c]) = cvt4(f);
  }
  asm volatile("" ::: "memory");
  layer_to_lds<192, 64>(Cat, W1, b1, sH1[wave], lane);
  layer_to_lds<64, 32>(sH1[wave], W2, b2, sH2[wave], lane);
  float outv[4 * 8];
  layer_final<32, 64>(sH2[wave], W3, b3, outv, lane);
  int n = lane & 15, half = lane >> 4;
#pragma unroll
  for (int nt = 0; nt < 4; ++nt)
#pragma unroll
    for (int r = 0; r < 8; ++r) {
      long g = r0 + r + half * 8;
      if (g < NGRAPHS) {
        int col = nt * 16 + n;
        out_states[g * 64 + col] = outv[nt * 8 + r] + states_pre[g * 64 + col];
      }
    }
}

extern "C" void kernel_launch(void* const* d_in, const int* in_sizes, int n_in,
                              void* d_out, int out_size, void* d_ws, size_t ws_size,
                              hipStream_t stream) {
  const float* sites  = (const float*)d_in[0];
  const float* bonds  = (const float*)d_in[1];
  const float* states = (const float*)d_in[2];
  const float* wf[16]; const float* bf[16];
  for (int i = 0; i < 15; ++i) { wf[i] = (const float*)d_in[3 + 2 * i]; bf[i] = (const float*)d_in[4 + 2 * i]; }
  // order: 0 sites_fc1, 1 sites_fc2, 2 bonds_fc1, 3 bonds_fc2, 4 states_fc1, 5 states_fc2,
  //        6 bu1, 7 bu2, 8 bu3, 9 su1, 10 su2, 11 su3, 12 stu1, 13 stu2, 14 stu3
  const int* indices1 = (const int*)d_in[33];
  const int* indices2 = (const int*)d_in[34];
  const int* graph_to_sites = (const int*)d_in[35];
  const int* graph_to_bonds = (const int*)d_in[36];

  float* out_sites  = (float*)d_out;
  float* out_bonds  = out_sites + (size_t)NSITES * DF;
  float* out_states = out_bonds + (size_t)NBONDS * DF;

  float* ws = (float*)d_ws;
  float* sites_pre  = ws;  ws += (size_t)NSITES * DF;
  float* bonds_pre  = ws;  ws += (size_t)NBONDS * DF;
  float* states_pre = ws;  ws += (size_t)NGRAPHS * DF;
  float* site_accum = ws;  ws += (size_t)NSITES * DF;
  float* site_cnt   = ws;  ws += NSITES;
  float* gb_accum   = ws;  ws += NGRAPHS * DF;
  float* gb_cnt     = ws;  ws += NGRAPHS;
  float* gs_accum   = ws;  ws += NGRAPHS * DF;
  float* gs_cnt     = ws;  ws += NGRAPHS;

  // packed bf16 weights (32B-aligned: all preceding sizes are multiples of 8 floats)
  static const int WK[15] = {64, 64, 64, 64, 64, 64, 256, 64, 32, 192, 64, 32, 192, 64, 32};
  static const int WM[15] = {64, 64, 64, 64, 64, 64, 64, 32, 64, 64, 32, 64, 64, 32, 64};
  __bf16* wp = (__bf16*)ws;
  __bf16* Wp[15];
  for (int i = 0; i < 15; ++i) { Wp[i] = wp; wp += (size_t)WK[i] * WM[i]; }
  // bf16 feature copies (gather sources; 16B aligned)
  __bf16* sites_bf  = wp;  wp += (size_t)NSITES * DF;
  __bf16* bonds_bf  = wp;  wp += (size_t)NBONDS * DF;
  __bf16* states_bf = wp;  wp += (size_t)NGRAPHS * DF;

  for (int i = 0; i < 15; ++i) {
    int tot = WK[i] * WM[i];
    pack_w_kernel<<<dim3((tot + 255) / 256), dim3(256), 0, stream>>>(wf[i], Wp[i], WK[i], WM[i]);
  }

  // zero all segment accumulators (contiguous from site_accum)
  long nz = (long)NSITES * DF + NSITES + 2L * (NGRAPHS * DF + NGRAPHS);
  zero_kernel<<<dim3((unsigned)((nz + 255) / 256)), dim3(256), 0, stream>>>(site_accum, nz);

  dim3 blk(128);  // 4 waves, 16 rows each -> 64 rows/block
  pre_mlp_kernel<<<dim3((NSITES / 16 + 3) / 4), blk, 0, stream>>>(
      sites, Wp[0], bf[0], Wp[1], bf[1], sites_pre, sites_bf, NSITES);
  pre_mlp_kernel<<<dim3((NBONDS / 16 + 3) / 4), blk, 0, stream>>>(
      bonds, Wp[2], bf[2], Wp[3], bf[3], bonds_pre, bonds_bf, NBONDS);
  pre_mlp_kernel<<<dim3(((NGRAPHS + 15) / 16 + 3) / 4), blk, 0, stream>>>(
      states, Wp[4], bf[4], Wp[5], bf[5], states_pre, states_bf, NGRAPHS);

  bond_update_kernel<<<dim3(NBONDS / 64), blk, 0, stream>>>(
      sites_bf, bonds_bf, states_bf, bonds_pre, indices1, indices2, graph_to_bonds,
      Wp[6], bf[6], Wp[7], bf[7], Wp[8], bf[8],
      out_bonds, site_accum, site_cnt, gb_accum, gb_cnt);

  site_update_kernel<<<dim3((NSITES / 16 + 3) / 4), blk, 0, stream>>>(
      sites_pre, sites_bf, states_bf, site_accum, site_cnt, graph_to_sites,
      Wp[9], bf[9], Wp[10], bf[10], Wp[11], bf[11],
      out_sites, gs_accum, gs_cnt);

  state_update_kernel<<<dim3(((NGRAPHS + 15) / 16 + 3) / 4), blk, 0, stream>>>(
      states_pre, gb_accum, gb_cnt, gs_accum, gs_cnt,
      Wp[12], bf[12], Wp[13], bf[13], Wp[14], bf[14],
      out_states);
}
